// GraphLayer_24008867185160
// MI455X (gfx1250) — matrix operand order repaired and checked by reading, not compiled, not run on hardware
//
#include <hip/hip_runtime.h>

// ---------------- problem constants ----------------
#define BB 2
#define NN 2048
#define DD 1024
#define HH 16
#define DHD 64
#define DII 4096
#define BN (BB * NN)      // 4096 rows
#define EPSV 1e-5f
#define NEG_SLOPE 0.01f

// ---------------- types ----------------
typedef __bf16 bf16_t;
typedef __attribute__((ext_vector_type(16))) __bf16 v16bf;
typedef __attribute__((ext_vector_type(8)))  float  v8f;
typedef __attribute__((ext_vector_type(4)))  float  v4f;
typedef __attribute__((ext_vector_type(4)))  unsigned int v4u;

union Frag16 { v16bf v; v4u u[2]; };

__device__ __forceinline__ bf16_t f2bf(float f) {
    unsigned u = __float_as_uint(f);
    unsigned r = (u + 0x7FFFu + ((u >> 16) & 1u)) >> 16;   // round-to-nearest-even
    unsigned short s = (unsigned short)r;
    return __builtin_bit_cast(bf16_t, s);
}

__device__ __forceinline__ v8f wmma_bf16(v16bf a, v16bf b, v8f c) {
    return __builtin_amdgcn_wmma_f32_16x16x32_bf16(false, a, false, b, (short)0, c, false, false);
}

__device__ __forceinline__ float bperm_f(int srcLane, float v) {
    return __int_as_float(__builtin_amdgcn_ds_bpermute(srcLane << 2, __float_as_int(v)));
}

__device__ __forceinline__ v16bf load_frag16(const bf16_t* p0, const bf16_t* p1) {
    Frag16 t;
    t.u[0] = *reinterpret_cast<const v4u*>(p0);
    t.u[1] = *reinterpret_cast<const v4u*>(p1);
    return t.v;
}

// ---------------- fp32 -> bf16 convert ----------------
__global__ __launch_bounds__(256) void convert_bf16_kernel(const float* __restrict__ src,
                                                           bf16_t* __restrict__ dst, int n) {
    int i = blockIdx.x * 256 + threadIdx.x;
    if (i < n) dst[i] = f2bf(src[i]);
}

// ---------------- LayerNorm: fp32 row -> bf16 row ----------------
__global__ __launch_bounds__(256) void ln_kernel(const float* __restrict__ X,
                                                 const float* __restrict__ g,
                                                 const float* __restrict__ b,
                                                 bf16_t* __restrict__ Y) {
    __shared__ float sbuf[512];
    const int t = threadIdx.x;
    const float* x = X + (size_t)blockIdx.x * DD;
    float x0 = x[t], x1 = x[t + 256], x2 = x[t + 512], x3 = x[t + 768];
    float s = x0 + x1 + x2 + x3;
    float q = x0 * x0 + x1 * x1 + x2 * x2 + x3 * x3;
    sbuf[t] = s; sbuf[256 + t] = q;
    __syncthreads();
    for (int st = 128; st > 0; st >>= 1) {
        if (t < st) { sbuf[t] += sbuf[t + st]; sbuf[256 + t] += sbuf[256 + t + st]; }
        __syncthreads();
    }
    const float mu  = sbuf[0]   * (1.f / DD);
    const float var = sbuf[256] * (1.f / DD) - mu * mu;
    const float rs  = rsqrtf(var + EPSV);
    bf16_t* y = Y + (size_t)blockIdx.x * DD;
    y[t]       = f2bf((x0 - mu) * rs * g[t]       + b[t]);
    y[t + 256] = f2bf((x1 - mu) * rs * g[t + 256] + b[t + 256]);
    y[t + 512] = f2bf((x2 - mu) * rs * g[t + 512] + b[t + 512]);
    y[t + 768] = f2bf((x3 - mu) * rs * g[t + 768] + b[t + 768]);
}

// ---------------- GEMM core: C[128x128 block] = A[M,K](bf16) * W[N,K]^T(bf16) ----------------
// 8 waves per block: wave (wm 0..3, wn 0..1), each wave 32(M) x 64(N) = 2x4 WMMA tiles.
__device__ __forceinline__ void gemm_core(const bf16_t* __restrict__ A,
                                          const bf16_t* __restrict__ W,
                                          int K, int rowbase, int colbase,
                                          v8f acc[2][4]) {
    const int lane = threadIdx.x & 31;
    const int nl = lane & 15, hi = lane >> 4;
    const int kb = hi * 16;
    const bf16_t* ar0 = A + (size_t)(rowbase + nl) * K;
    const bf16_t* ar1 = A + (size_t)(rowbase + 16 + nl) * K;
    const bf16_t* wr0 = W + (size_t)(colbase +  0 + nl) * K + hi * 16;
    const bf16_t* wr1 = W + (size_t)(colbase + 16 + nl) * K + hi * 16;
    const bf16_t* wr2 = W + (size_t)(colbase + 32 + nl) * K + hi * 16;
    const bf16_t* wr3 = W + (size_t)(colbase + 48 + nl) * K + hi * 16;
    for (int k0 = 0; k0 < K; k0 += 32) {
        v16bf af0 = load_frag16(ar0 + k0 + kb, ar0 + k0 + kb + 8);
        v16bf af1 = load_frag16(ar1 + k0 + kb, ar1 + k0 + kb + 8);
        v16bf bf0 = load_frag16(wr0 + k0, wr0 + k0 + 8);
        v16bf bf1 = load_frag16(wr1 + k0, wr1 + k0 + 8);
        v16bf bf2 = load_frag16(wr2 + k0, wr2 + k0 + 8);
        v16bf bf3 = load_frag16(wr3 + k0, wr3 + k0 + 8);
        acc[0][0] = wmma_bf16(af0, bf0, acc[0][0]);
        acc[0][1] = wmma_bf16(af0, bf1, acc[0][1]);
        acc[0][2] = wmma_bf16(af0, bf2, acc[0][2]);
        acc[0][3] = wmma_bf16(af0, bf3, acc[0][3]);
        acc[1][0] = wmma_bf16(af1, bf0, acc[1][0]);
        acc[1][1] = wmma_bf16(af1, bf1, acc[1][1]);
        acc[1][2] = wmma_bf16(af1, bf2, acc[1][2]);
        acc[1][3] = wmma_bf16(af1, bf3, acc[1][3]);
    }
}

// ---------------- QKV GEMM with scatter epilogue ----------------
// out col c in [0,3072): h = c/192, j = c%192; j<64 -> V (transposed per head),
// 64<=j<128 -> Q (row-major per head), j>=128 -> K (row-major per head). chunk order: V,Q,K.
__global__ __launch_bounds__(256) void gemm_qkv_kernel(const bf16_t* __restrict__ A,
                                                       const bf16_t* __restrict__ W,
                                                       const float* __restrict__ bias,
                                                       bf16_t* __restrict__ Qh,
                                                       bf16_t* __restrict__ Kh,
                                                       bf16_t* __restrict__ Vt) {
    const int w = threadIdx.x >> 5;
    const int rowbase = blockIdx.y * 128 + (w & 3) * 32;
    const int colbase = blockIdx.x * 128 + (w >> 2) * 64;
    v8f acc[2][4] = {};
    gemm_core(A, W, DD, rowbase, colbase, acc);
    const int lane = threadIdx.x & 31;
    const int nl = lane & 15, hi = lane >> 4;
    for (int ni = 0; ni < 4; ni++) {
        const int col = colbase + ni * 16 + nl;
        const int h = col / (3 * DHD);
        const int j = col - h * (3 * DHD);
        const float bv = bias[col];
        for (int mi = 0; mi < 2; mi++) {
            for (int v = 0; v < 8; v++) {
                const int row = rowbase + mi * 16 + v + 8 * hi;       // 0..4095
                const int bb = row >> 11, n = row & (NN - 1);
                const int bh = bb * HH + h;
                const bf16_t val = f2bf(acc[mi][ni][v] + bv);
                if (j < DHD) {
                    Vt[((size_t)bh * DHD + j) * NN + n] = val;        // Vt[bh][d][m]
                } else if (j < 2 * DHD) {
                    Qh[((size_t)bh * NN + n) * DHD + (j - DHD)] = val;
                } else {
                    Kh[((size_t)bh * NN + n) * DHD + (j - 2 * DHD)] = val;
                }
            }
        }
    }
}

// ---------------- generic GEMM epilogues ----------------
// EPI 0: out_f32 = acc + resid              (o-proj + residual -> Z1)
// EPI 1: out_bf16 = relu(acc + bias)        (MLP layer 1 -> h)
// EPI 2: out_f32 = acc + bias + resid       (MLP layer 2 + residual -> final)
template <int EPI>
__global__ __launch_bounds__(256) void gemm_bf16_kernel(const bf16_t* __restrict__ A,
                                                        const bf16_t* __restrict__ W,
                                                        int K, int Ncols,
                                                        const float* __restrict__ bias,
                                                        const float* __restrict__ resid,
                                                        float* __restrict__ outf,
                                                        bf16_t* __restrict__ outb) {
    const int w = threadIdx.x >> 5;
    const int rowbase = blockIdx.y * 128 + (w & 3) * 32;
    const int colbase = blockIdx.x * 128 + (w >> 2) * 64;
    v8f acc[2][4] = {};
    gemm_core(A, W, K, rowbase, colbase, acc);
    const int lane = threadIdx.x & 31;
    const int nl = lane & 15, hi = lane >> 4;
    for (int ni = 0; ni < 4; ni++) {
        const int col = colbase + ni * 16 + nl;
        const float bv = (EPI == 0) ? 0.f : bias[col];
        for (int mi = 0; mi < 2; mi++) {
            for (int v = 0; v < 8; v++) {
                const int row = rowbase + mi * 16 + v + 8 * hi;
                const size_t idx = (size_t)row * Ncols + col;
                float val = acc[mi][ni][v] + bv;
                if (EPI == 0) {
                    outf[idx] = val + resid[idx];
                } else if (EPI == 1) {
                    outb[idx] = f2bf(val > 0.f ? val : 0.f);
                } else {
                    outf[idx] = val + resid[idx];
                }
            }
        }
    }
}

// ---------------- flash attention ----------------
// grid: (N/128, B*H). 8 independent waves/block, each owns 16 query rows.
// S^T tiles (m x n) via WMMA(A=K, B=Q^T); online softmax stats per lane (n = lane%16,
// duplicated across halves); P^T C-layout maps directly to the A-fragment of P for P@V.
__global__ __launch_bounds__(256) void flash_kernel(const bf16_t* __restrict__ Qh,
                                                    const bf16_t* __restrict__ Kh,
                                                    const bf16_t* __restrict__ Vt,
                                                    const float* __restrict__ nmask,
                                                    const float* __restrict__ mmask,
                                                    bf16_t* __restrict__ attn_lr) {
    const int lane = threadIdx.x & 31;
    const int w = threadIdx.x >> 5;
    const int nl = lane & 15, hi = lane >> 4;
    const int bh = blockIdx.y, bb = bh >> 4, h = bh & 15;
    const int n0 = blockIdx.x * 128 + w * 16;

    const bf16_t* Qb = Qh + ((size_t)bh * NN + n0) * DHD;
    const bf16_t* Kb = Kh + (size_t)bh * NN * DHD;
    const bf16_t* Vb = Vt + (size_t)bh * DHD * NN;
    const float* nmb = nmask + ((size_t)bb * NN + (n0 + nl)) * NN;
    const float* mkb = mmask + ((size_t)bb * NN + (n0 + nl)) * NN;

    // Q^T B-fragments (d-chunks of 32); lane: n=nl, K(d) offset 16*hi
    const bf16_t* qp = Qb + (size_t)nl * DHD + hi * 16;
    v16bf qf0 = load_frag16(qp,      qp + 8);        // d 0..31
    v16bf qf1 = load_frag16(qp + 32, qp + 40);       // d 32..63

    v8f acc[4] = {};
    float mrun = -3.0e38f, lsum = 0.f;

    for (int m0 = 0; m0 < NN; m0 += 32) {
        // ---- S^T tiles (two 16-m tiles) ----
        v8f st[2];
        for (int t = 0; t < 2; t++) {
            const bf16_t* kp = Kb + (size_t)(m0 + t * 16 + nl) * DHD + hi * 16;
            v16bf kf0 = load_frag16(kp,      kp + 8);       // d 0..31 chunk
            v16bf kf1 = load_frag16(kp + 32, kp + 40);      // d 32..63 chunk
            v8f s = {};
            s = wmma_bf16(kf0, qf0, s);
            s = wmma_bf16(kf1, qf1, s);
            st[t] = s;
        }
        // ---- scale + new_mask, block max ----
        float sv[2][8], mkv[2][8];
        float bm = -3.0e38f;
        for (int t = 0; t < 2; t++) {
            const int moff = m0 + t * 16 + 8 * hi;
            v4f a0 = *reinterpret_cast<const v4f*>(nmb + moff);
            v4f a1 = *reinterpret_cast<const v4f*>(nmb + moff + 4);
            v4f k0 = *reinterpret_cast<const v4f*>(mkb + moff);
            v4f k1 = *reinterpret_cast<const v4f*>(mkb + moff + 4);
            for (int v = 0; v < 4; v++) {
                sv[t][v]     = st[t][v]     * 0.125f + a0[v];
                sv[t][v + 4] = st[t][v + 4] * 0.125f + a1[v];
                mkv[t][v] = k0[v]; mkv[t][v + 4] = k1[v];
            }
            for (int v = 0; v < 8; v++) bm = fmaxf(bm, sv[t][v]);
        }
        bm = fmaxf(bm, bperm_f(lane ^ 16, bm));                 // combine lane halves
        const float mnew  = fmaxf(mrun, bm);
        const float scale = __expf(mrun - mnew);
        mrun = mnew;
        // ---- P = exp(s-m)*mask, pack directly into A-fragment of P ----
        float ls = 0.f;
        v16bf pf;
        for (int t = 0; t < 2; t++) {
            for (int v = 0; v < 8; v++) {
                const float e = __expf(sv[t][v] - mnew);
                ls += e;                                        // denominator unmasked
                pf[t * 8 + v] = f2bf(e * mkv[t][v]);
            }
        }
        ls += bperm_f(lane ^ 16, ls);
        lsum = lsum * scale + ls;
        // ---- rescale accumulators (per output row n = v + 8*hi) ----
        float sc[8];
        for (int v = 0; v < 8; v++) sc[v] = bperm_f(8 * hi + v, scale);
        for (int dt = 0; dt < 4; dt++)
            for (int v = 0; v < 8; v++) acc[dt][v] *= sc[v];
        // ---- acc += P @ V ----
        for (int dt = 0; dt < 4; dt++) {
            const bf16_t* vp = Vb + (size_t)(dt * 16 + nl) * NN + m0 + hi * 8;
            v16bf vf = load_frag16(vp, vp + 16);
            acc[dt] = wmma_bf16(pf, vf, acc[dt]);
        }
    }
    // ---- epilogue: /l, leaky-ReLU, bf16 store ----
    float linv[8];
    for (int v = 0; v < 8; v++) linv[v] = 1.f / bperm_f(8 * hi + v, lsum);
    for (int dt = 0; dt < 4; dt++) {
        for (int v = 0; v < 8; v++) {
            float o = acc[dt][v] * linv[v];
            o = (o >= 0.f) ? o : NEG_SLOPE * o;
            const int row = n0 + v + 8 * hi;                    // query n
            attn_lr[((size_t)bb * NN + row) * DD + h * DHD + dt * 16 + nl] = f2bf(o);
        }
    }
}

// ---------------- host launcher ----------------
extern "C" void kernel_launch(void* const* d_in, const int* in_sizes, int n_in,
                              void* d_out, int out_size, void* d_ws, size_t ws_size,
                              hipStream_t stream) {
    const float* Z      = (const float*)d_in[0];
    const float* nmask  = (const float*)d_in[1];
    const float* mmask  = (const float*)d_in[2];
    const float* ln1_g  = (const float*)d_in[3];
    const float* ln1_b  = (const float*)d_in[4];
    const float* qkv_w  = (const float*)d_in[5];
    const float* qkv_b  = (const float*)d_in[6];
    const float* o_w    = (const float*)d_in[7];
    const float* ln2_g  = (const float*)d_in[8];
    const float* ln2_b  = (const float*)d_in[9];
    const float* p1_w   = (const float*)d_in[10];
    const float* p1_b   = (const float*)d_in[11];
    const float* p2_w   = (const float*)d_in[12];
    const float* p2_b   = (const float*)d_in[13];
    float* out = (float*)d_out;

    char* ws = (char*)d_ws;
    size_t off = 0;
    auto alloc = [&](size_t bytes) -> void* {
        void* p = ws + off;
        off += (bytes + 255) & ~(size_t)255;
        return p;
    };
    bf16_t* Zn      = (bf16_t*)alloc((size_t)BN * DD * 2);
    bf16_t* qkvw_b  = (bf16_t*)alloc((size_t)3 * DD * DD * 2);
    bf16_t* Qh      = (bf16_t*)alloc((size_t)BB * HH * NN * DHD * 2);
    bf16_t* Kh      = (bf16_t*)alloc((size_t)BB * HH * NN * DHD * 2);
    bf16_t* Vt      = (bf16_t*)alloc((size_t)BB * HH * NN * DHD * 2);
    bf16_t* attn_lr = (bf16_t*)alloc((size_t)BN * DD * 2);
    bf16_t* ow_b    = (bf16_t*)alloc((size_t)DD * DD * 2);
    float*  Z1      = (float*) alloc((size_t)BN * DD * 4);
    bf16_t* Zn2     = (bf16_t*)alloc((size_t)BN * DD * 2);
    bf16_t* p1w_b   = (bf16_t*)alloc((size_t)DII * DD * 2);
    bf16_t* hbuf    = (bf16_t*)alloc((size_t)BN * DII * 2);
    bf16_t* p2w_b   = (bf16_t*)alloc((size_t)DD * DII * 2);

    // weight conversion fp32 -> bf16
    convert_bf16_kernel<<<(3 * DD * DD) / 256, 256, 0, stream>>>(qkv_w, qkvw_b, 3 * DD * DD);
    convert_bf16_kernel<<<(DD * DD) / 256, 256, 0, stream>>>(o_w, ow_b, DD * DD);
    convert_bf16_kernel<<<(DII * DD) / 256, 256, 0, stream>>>(p1_w, p1w_b, DII * DD);
    convert_bf16_kernel<<<(DD * DII) / 256, 256, 0, stream>>>(p2_w, p2w_b, DD * DII);

    // LN1
    ln_kernel<<<BN, 256, 0, stream>>>(Z, ln1_g, ln1_b, Zn);

    // QKV projection (scatter to Q/K row-major, V transposed, bf16)
    gemm_qkv_kernel<<<dim3((3 * DD) / 128, BN / 128), 256, 0, stream>>>(
        Zn, qkvw_b, qkv_b, Qh, Kh, Vt);

    // flash attention + leaky ReLU
    flash_kernel<<<dim3(NN / 128, BB * HH), 256, 0, stream>>>(
        Qh, Kh, Vt, nmask, mmask, attn_lr);

    // output projection + residual -> Z1 (fp32)
    gemm_bf16_kernel<0><<<dim3(DD / 128, BN / 128), 256, 0, stream>>>(
        attn_lr, ow_b, DD, DD, nullptr, Z, Z1, nullptr);

    // LN2
    ln_kernel<<<BN, 256, 0, stream>>>(Z1, ln2_g, ln2_b, Zn2);

    // MLP layer 1 (+bias, ReLU) -> h (bf16)
    gemm_bf16_kernel<1><<<dim3(DII / 128, BN / 128), 256, 0, stream>>>(
        Zn2, p1w_b, DD, DII, p1_b, nullptr, nullptr, hbuf);

    // MLP layer 2 (+bias +Z1 residual) -> out (fp32)
    gemm_bf16_kernel<2><<<dim3(DD / 128, BN / 128), 256, 0, stream>>>(
        hbuf, p2w_b, DII, DD, p2_b, Z1, out, nullptr);
}
